// MultiHeadAttention_4372276707345
// MI455X (gfx1250) — matrix operand order repaired and checked by reading, not compiled, not run on hardware
//
#include <hip/hip_runtime.h>
#include <hip/hip_bf16.h>

// ---------------------------------------------------------------------------
// MultiHeadAttention forward for MI455X (gfx1250, wave32, WMMA).
// All GEMMs use v_wmma_f32_16x16x32_bf16 (bf16 A/B, fp32 accum).
// Attention is flash-style: scores never touch HBM; K tiles are staged into
// LDS with the CDNA5 async-to-LDS engine (ASYNCcnt), double buffered.
// Large GEMMs use 32x64 tiles per wave (8 WMMAs / 6 fragment loads).
// ---------------------------------------------------------------------------

typedef __attribute__((ext_vector_type(16))) __bf16 v16bf;
typedef __attribute__((ext_vector_type(8)))  float  v8f;

constexpr int NB  = 4;
constexpr int NSQ = 2048;
constexpr int NSK = 2048;
constexpr int ND  = 1024;
constexpr int NH  = 16;
constexpr int NDK = 64;
constexpr int NDV = 64;

// sched_group_barrier needs immediate args -> template parameters.
template <int NLOADS, int NWMMA>
__device__ __forceinline__ void sched_loads_then_wmma() {
#if __has_builtin(__builtin_amdgcn_sched_group_barrier)
  __builtin_amdgcn_sched_group_barrier(0x20, NLOADS, 0);  // VMEM reads first
  __builtin_amdgcn_sched_group_barrier(0x08, NWMMA, 0);   // then WMMA burst
#endif
}

// ---------------------------------------------------------------------------
// WMMA wrapper: D = A(16x32 bf16) * B(32x16 bf16) + C(16x16 f32)
// ---------------------------------------------------------------------------
__device__ __forceinline__ v8f wmma_bf16(v16bf a, v16bf b, v8f c) {
  return __builtin_amdgcn_wmma_f32_16x16x32_bf16(
      /*neg_a=*/false, a, /*neg_b=*/false, b,
      /*c_mod=*/(short)0, c, /*reuse_a=*/false, /*reuse_b=*/false);
}

// A-fragment (16x32, MxK) from row-major bf16 [.., lda]:
//   lane<16 : row=lane,    elems 0..7 = K 0..7,  8..15 = K 16..23
//   lane>=16: row=lane-16, elems 0..7 = K 8..15, 8..15 = K 24..31
__device__ __forceinline__ v16bf load_a_bf16(const __bf16* A, int lda,
                                             int row0, int k0) {
  const int lane = threadIdx.x & 31;
  const __bf16* p = A + (size_t)(row0 + (lane & 15)) * lda + k0 + ((lane >> 4) << 3);
  v16bf a;
#pragma unroll
  for (int i = 0; i < 8; ++i) { a[i] = p[i]; a[i + 8] = p[16 + i]; }
  return a;
}

// Same A-fragment but sourced from fp32 row-major (convert-on-load).
__device__ __forceinline__ v16bf load_a_f32(const float* A, int lda,
                                            int row0, int k0) {
  const int lane = threadIdx.x & 31;
  const float* p = A + (size_t)(row0 + (lane & 15)) * lda + k0 + ((lane >> 4) << 3);
  v16bf a;
#pragma unroll
  for (int i = 0; i < 8; ++i) {
    a[i]     = (__bf16)p[i];
    a[i + 8] = (__bf16)p[16 + i];
  }
  return a;
}

// B-fragment (32x16, KxN) from "Bt" stored N-major ([N rows, K cols], ldb):
//   lane = column N (lane&15); same K order as A: lanes 0-15 K 0..7,16..23; lanes 16-31 K 8..15,24..31.
__device__ __forceinline__ v16bf load_b_bf16(const __bf16* Bt, int ldb,
                                             int n0, int k0) {
  const int lane = threadIdx.x & 31;
  const __bf16* p = Bt + (size_t)(n0 + (lane & 15)) * ldb + k0 + ((lane >> 4) << 3);
  v16bf b;
#pragma unroll
  for (int i = 0; i < 8; ++i) { b[i] = p[i]; b[i + 8] = p[16 + i]; }
  return b;
}

// ---------------------------------------------------------------------------
// Weight transpose + f32->bf16 convert:  in[batch][R][C] f32 -> out[batch][C][R] bf16
// ---------------------------------------------------------------------------
__global__ void k_tcvt_f32(const float* __restrict__ in, __bf16* __restrict__ out,
                           int R, int C, size_t total) {
  const size_t rc = (size_t)R * C;
  for (size_t i = (size_t)blockIdx.x * blockDim.x + threadIdx.x; i < total;
       i += (size_t)gridDim.x * blockDim.x) {
    size_t bidx = i / rc, rem = i % rc;
    size_t r = rem / C, c = rem % C;
    out[bidx * rc + c * R + r] = (__bf16)in[i];
  }
}

// bf16 transpose: in[batch][R][C] -> out[batch][C][R]  (vh -> vhT)
__global__ void k_t_bf16(const __bf16* __restrict__ in, __bf16* __restrict__ out,
                         int R, int C, size_t total) {
  const size_t rc = (size_t)R * C;
  for (size_t i = (size_t)blockIdx.x * blockDim.x + threadIdx.x; i < total;
       i += (size_t)gridDim.x * blockDim.x) {
    size_t bidx = i / rc, rem = i % rc;
    size_t r = rem / C, c = rem % C;
    out[bidx * rc + c * R + r] = in[i];
  }
}

// ---------------------------------------------------------------------------
// QKV projection: Y[b,h,s,:] = X[b,s,:] @ W[h] + bias[h]
//   32x64 tile per wave: 2 A-frags x 4 B-frags -> 8 WMMAs per K step.
//   launch_bounds(128,1): allow full register allocation (no acc spills).
// ---------------------------------------------------------------------------
__global__ __launch_bounds__(128, 1)
void k_proj(const float* __restrict__ X, const __bf16* __restrict__ WT,
            const float* __restrict__ bias, __bf16* __restrict__ Y,
            int S) {
  const int mtiles = S / 128;  // block covers 128 rows (4 waves x 32)
  int id = blockIdx.x;
  const int mt = id % mtiles; id /= mtiles;
  const int h  = id % NH;
  const int b  = id / NH;
  const int wave = threadIdx.x >> 5, lane = threadIdx.x & 31;

  const float*  A  = X + (size_t)b * S * ND;
  const __bf16* Bt = WT + (size_t)h * NDK * ND;
  const float*  bs = bias + h * NDK;
  __bf16*       C  = Y + ((size_t)(b * NH + h) * S) * NDK;

  const int m0 = mt * 128 + wave * 32;
  v8f acc[8] = {};
  for (int k0 = 0; k0 < ND; k0 += 32) {
    if (k0 + 32 < ND)  // next A tile -> global_prefetch_b8
      __builtin_prefetch(A + (size_t)(m0 + (lane & 15)) * ND + k0 + 32, 0, 1);
    v16bf a0 = load_a_f32(A, ND, m0, k0);
    v16bf a1 = load_a_f32(A, ND, m0 + 16, k0);
    v16bf b0 = load_b_bf16(Bt, ND, 0,  k0);
    v16bf b1 = load_b_bf16(Bt, ND, 16, k0);
    v16bf b2 = load_b_bf16(Bt, ND, 32, k0);
    v16bf b3 = load_b_bf16(Bt, ND, 48, k0);
    acc[0] = wmma_bf16(a0, b0, acc[0]);
    acc[1] = wmma_bf16(a0, b1, acc[1]);
    acc[2] = wmma_bf16(a0, b2, acc[2]);
    acc[3] = wmma_bf16(a0, b3, acc[3]);
    acc[4] = wmma_bf16(a1, b0, acc[4]);
    acc[5] = wmma_bf16(a1, b1, acc[5]);
    acc[6] = wmma_bf16(a1, b2, acc[6]);
    acc[7] = wmma_bf16(a1, b3, acc[7]);
    sched_loads_then_wmma<16, 8>();
  }
  const int ncol = lane & 15, mb = (lane >> 4) << 3;
#pragma unroll
  for (int f = 0; f < 4; ++f) {
    const float bv = bs[f * 16 + ncol];
#pragma unroll
    for (int r = 0; r < 8; ++r) {
      C[(size_t)(m0 + mb + r) * NDK + f * 16 + ncol]      = (__bf16)(acc[f][r] + bv);
      C[(size_t)(m0 + 16 + mb + r) * NDK + f * 16 + ncol] = (__bf16)(acc[4 + f][r] + bv);
    }
  }
}

// ---------------------------------------------------------------------------
// Flash attention: one wave per (b, h, 16-query tile).
//   qh : bf16 [B,H,SQ,64]  kh : bf16 [B,H,SK,64]  vT : bf16 [B,H,64,SK]
//   K tiles (32 keys x 64 = 4KB, contiguous) staged to LDS with
//   global_load_async_to_lds_b128, double buffered on ASYNCcnt.
// ---------------------------------------------------------------------------
__global__ __launch_bounds__(128, 1)
void k_flash_attn(const __bf16* __restrict__ qh,
                  const __bf16* __restrict__ kh,
                  const __bf16* __restrict__ vT,
                  const unsigned char* __restrict__ pad,
                  __bf16* __restrict__ ctx) {
  __shared__ __align__(16) __bf16 kstage[4][2][32 * NDK];  // 32 KB: per-wave dbl buf
  __shared__ __align__(16) __bf16 pbuf[4][16 * 32];        //  4 KB: P re-layout stage

  const int wave = threadIdx.x >> 5, lane = threadIdx.x & 31;
  int gw = blockIdx.x * 4 + wave;
  const int qt = gw % (NSQ / 16); gw /= (NSQ / 16);
  const int h  = gw % NH;
  const int b  = gw / NH;

  const __bf16* q_bh = qh + ((size_t)(b * NH + h)) * NSQ * NDK;
  const __bf16* k_bh = kh + ((size_t)(b * NH + h)) * NSK * NDK;
  const __bf16* v_bh = vT + ((size_t)(b * NH + h)) * NDV * NSK;
  const unsigned char* pad_b = pad + (size_t)b * NSK;

  // LDS byte offsets for the async engine (low 32 bits of the flat address)
  const unsigned ks_lds0 = (unsigned)(uintptr_t)&kstage[wave][0][0];
  const unsigned ks_lds1 = (unsigned)(uintptr_t)&kstage[wave][1][0];

  // Stage one 4KB K tile: 8 issues x (32 lanes x 16B); INST_OFFSET advances
  // both the global and the LDS address (ISA 10.7 async addressing).
  auto stage_k = [&](int kt, int buf) {
    const unsigned lds_addr =
        (buf ? ks_lds1 : ks_lds0) + (unsigned)(lane * 16);
    const unsigned long long ga =
        (unsigned long long)(uintptr_t)(k_bh + (size_t)kt * 32 * NDK) +
        (unsigned)(lane * 16);
    asm volatile(
        "global_load_async_to_lds_b128 %0, %1, off\n\t"
        "global_load_async_to_lds_b128 %0, %1, off offset:512\n\t"
        "global_load_async_to_lds_b128 %0, %1, off offset:1024\n\t"
        "global_load_async_to_lds_b128 %0, %1, off offset:1536\n\t"
        "global_load_async_to_lds_b128 %0, %1, off offset:2048\n\t"
        "global_load_async_to_lds_b128 %0, %1, off offset:2560\n\t"
        "global_load_async_to_lds_b128 %0, %1, off offset:3072\n\t"
        "global_load_async_to_lds_b128 %0, %1, off offset:3584"
        :: "v"(lds_addr), "v"(ga) : "memory");
  };

  // Q fragments (reused across all key tiles)
  const v16bf aq0 = load_a_bf16(q_bh, NDK, qt * 16, 0);
  const v16bf aq1 = load_a_bf16(q_bh, NDK, qt * 16, 32);

  v8f acc0 = {}, acc1 = {}, acc2 = {}, acc3 = {};
  float m[8], lsum[8];
#pragma unroll
  for (int r = 0; r < 8; ++r) { m[r] = -__builtin_inff(); lsum[r] = 0.f; }

  const int mrow_base = (lane >> 4) << 3;  // rows 0..7 or 8..15
  const int ncol = lane & 15;
  const int q_last = qt * 16 + 15;
  const int nkt = (q_last >> 5) + 1;       // causal: 32-key tiles needed

  stage_k(0, 0);  // prologue fill

  for (int kt = 0; kt < nkt; ++kt) {
    const int kbase = kt * 32;
    const __bf16* kcur = &kstage[wave][kt & 1][0];

    if (kt + 1 < nkt) {
      // previous reads of the other buffer must be done before overwrite
      asm volatile("s_wait_dscnt 0x0" ::: "memory");
      stage_k(kt + 1, (kt + 1) & 1);
      // 8 ops now in flight for kt+1; <=8 outstanding => kt's tile landed
      asm volatile("s_wait_asynccnt 0x8" ::: "memory");
    } else {
      asm volatile("s_wait_asynccnt 0x0" ::: "memory");
    }

    // ---- S = Q K^T for 16 queries x 32 keys (4 WMMAs, K from LDS) -------
    v16bf kb00 = load_b_bf16(kcur, NDK, 0,  0);
    v16bf kb01 = load_b_bf16(kcur, NDK, 0,  32);
    v16bf kb10 = load_b_bf16(kcur, NDK, 16, 0);
    v16bf kb11 = load_b_bf16(kcur, NDK, 16, 32);
    v8f s0 = {}, s1 = {};
    s0 = wmma_bf16(aq0, kb00, s0);
    s0 = wmma_bf16(aq1, kb01, s0);
    s1 = wmma_bf16(aq0, kb10, s1);
    s1 = wmma_bf16(aq1, kb11, s1);

    // ---- scale + causal + padding mask (in C layout) --------------------
    const int key0 = kbase + ncol, key1 = kbase + 16 + ncol;
    const float pm0 = pad_b[key0] ? -__builtin_inff() : 0.f;
    const float pm1 = pad_b[key1] ? -__builtin_inff() : 0.f;
#pragma unroll
    for (int r = 0; r < 8; ++r) {
      const int qrow = qt * 16 + mrow_base + r;
      float v0 = s0[r] * 0.125f + pm0;  // 1/sqrt(64)
      float v1 = s1[r] * 0.125f + pm1;
      if (key0 > qrow) v0 = -__builtin_inff();
      if (key1 > qrow) v1 = -__builtin_inff();
      s0[r] = v0; s1[r] = v1;
    }

    // ---- online softmax: rowwise max/sum across 16-lane halves ----------
    float corr[8];
#pragma unroll
    for (int r = 0; r < 8; ++r) {
      float tmax = fmaxf(s0[r], s1[r]);
#pragma unroll
      for (int off = 1; off < 16; off <<= 1)
        tmax = fmaxf(tmax, __shfl_xor(tmax, off, 32));
      const float mnew = fmaxf(m[r], tmax);
      const float p0 = __expf(s0[r] - mnew);
      const float p1 = __expf(s1[r] - mnew);
      float tsum = p0 + p1;
#pragma unroll
      for (int off = 1; off < 16; off <<= 1)
        tsum += __shfl_xor(tsum, off, 32);
      corr[r] = __expf(m[r] - mnew);
      lsum[r] = lsum[r] * corr[r] + tsum;
      m[r] = mnew;
      s0[r] = p0; s1[r] = p1;
    }
#pragma unroll
    for (int r = 0; r < 8; ++r) {
      acc0[r] *= corr[r]; acc1[r] *= corr[r];
      acc2[r] *= corr[r]; acc3[r] *= corr[r];
    }

    // ---- re-layout P (C layout -> A layout) via per-wave LDS tile -------
    __bf16* pb = pbuf[wave];
#pragma unroll
    for (int r = 0; r < 8; ++r) {
      pb[(mrow_base + r) * 32 + ncol]      = (__bf16)s0[r];
      pb[(mrow_base + r) * 32 + 16 + ncol] = (__bf16)s1[r];
    }
    asm volatile("" ::: "memory");  // same-wave DS ops are in order
    v16bf ap;
    {
      const __bf16* pr = pb + (lane & 15) * 32 + ((lane >> 4) << 3);
#pragma unroll
      for (int i = 0; i < 8; ++i) { ap[i] = pr[i]; ap[i + 8] = pr[16 + i]; }
    }
    asm volatile("" ::: "memory");

    // ---- acc += P (16x32) x V (32x64): V B-frags contiguous from vhT ----
    v16bf vb0 = load_b_bf16(v_bh, NSK, 0,  kbase);
    v16bf vb1 = load_b_bf16(v_bh, NSK, 16, kbase);
    v16bf vb2 = load_b_bf16(v_bh, NSK, 32, kbase);
    v16bf vb3 = load_b_bf16(v_bh, NSK, 48, kbase);
    acc0 = wmma_bf16(ap, vb0, acc0);
    acc1 = wmma_bf16(ap, vb1, acc1);
    acc2 = wmma_bf16(ap, vb2, acc2);
    acc3 = wmma_bf16(ap, vb3, acc3);
  }

  // ---- epilogue: normalize (with nan_to_num guard) and store ctx --------
  float inv[8];
#pragma unroll
  for (int r = 0; r < 8; ++r) inv[r] = (lsum[r] > 0.f) ? 1.f / lsum[r] : 0.f;
  __bf16* c_bh = ctx + ((size_t)(b * NH + h) * NSQ + qt * 16) * NDV;
  v8f* accs[4] = {&acc0, &acc1, &acc2, &acc3};
#pragma unroll
  for (int f = 0; f < 4; ++f) {
#pragma unroll
    for (int r = 0; r < 8; ++r)
      c_bh[(size_t)(mrow_base + r) * NDV + f * 16 + ncol] =
          (__bf16)((*accs[f])[r] * inv[r]);
  }
}

// ---------------------------------------------------------------------------
// Per-head output projection into concat layout:
//   heads[b,s, h*64+n] = ctx[b,h,s,:] @ Wo[h] + bo[h]
// ---------------------------------------------------------------------------
__global__ void k_wo(const __bf16* __restrict__ CTX, const __bf16* __restrict__ WoT,
                     const float* __restrict__ bo, __bf16* __restrict__ HEADS) {
  const int mtiles = NSQ / 64;
  int id = blockIdx.x;
  const int mt = id % mtiles; id /= mtiles;
  const int h  = id % NH;
  const int b  = id / NH;
  const int wave = threadIdx.x >> 5, lane = threadIdx.x & 31;

  const __bf16* A  = CTX + ((size_t)(b * NH + h)) * NSQ * NDV;
  const __bf16* Bt = WoT + (size_t)h * NDV * NDV;
  const float*  bs = bo + h * NDV;

  const int m0 = mt * 64 + wave * 16;
  v8f acc[4] = {};
#pragma unroll
  for (int k0 = 0; k0 < NDV; k0 += 32) {
    v16bf a  = load_a_bf16(A, NDV, m0, k0);
    v16bf b0 = load_b_bf16(Bt, NDV, 0,  k0);
    v16bf b1 = load_b_bf16(Bt, NDV, 16, k0);
    v16bf b2 = load_b_bf16(Bt, NDV, 32, k0);
    v16bf b3 = load_b_bf16(Bt, NDV, 48, k0);
    acc[0] = wmma_bf16(a, b0, acc[0]);
    acc[1] = wmma_bf16(a, b1, acc[1]);
    acc[2] = wmma_bf16(a, b2, acc[2]);
    acc[3] = wmma_bf16(a, b3, acc[3]);
  }
  const int ncol = lane & 15, mb = (lane >> 4) << 3;
#pragma unroll
  for (int f = 0; f < 4; ++f) {
    const float bv = bs[f * 16 + ncol];
#pragma unroll
    for (int r = 0; r < 8; ++r)
      HEADS[((size_t)b * NSQ + m0 + mb + r) * (NH * NDV) + h * NDV + f * 16 + ncol] =
          (__bf16)(acc[f][r] + bv);
  }
}

// ---------------------------------------------------------------------------
// Final projection: out = concat(heads) @ Wf + bf   (1024x1024, fp32 out)
//   32x64 tile per wave, same shape as k_proj.
// ---------------------------------------------------------------------------
__global__ __launch_bounds__(128, 1)
void k_final(const __bf16* __restrict__ HEADS, const __bf16* __restrict__ WfT,
             const float* __restrict__ bfin, float* __restrict__ OUT) {
  const int ntiles = ND / 64;        // 16
  const int mtiles = NSQ / 128;      // 16 (128 rows per block)
  int id = blockIdx.x;
  const int nt = id % ntiles; id /= ntiles;
  const int mt = id % mtiles;
  const int b  = id / mtiles;
  const int wave = threadIdx.x >> 5, lane = threadIdx.x & 31;

  const __bf16* A = HEADS + (size_t)b * NSQ * (NH * NDV);
  const int m0 = mt * 128 + wave * 32;
  const int HD = NH * NDV;  // 1024

  v8f acc[8] = {};
  for (int k0 = 0; k0 < HD; k0 += 32) {
    if (k0 + 32 < HD)
      __builtin_prefetch(A + (size_t)(m0 + (lane & 15)) * HD + k0 + 32, 0, 1);
    v16bf a0 = load_a_bf16(A, HD, m0, k0);
    v16bf a1 = load_a_bf16(A, HD, m0 + 16, k0);
    v16bf b0 = load_b_bf16(WfT, HD, nt * 64 + 0,  k0);
    v16bf b1 = load_b_bf16(WfT, HD, nt * 64 + 16, k0);
    v16bf b2 = load_b_bf16(WfT, HD, nt * 64 + 32, k0);
    v16bf b3 = load_b_bf16(WfT, HD, nt * 64 + 48, k0);
    acc[0] = wmma_bf16(a0, b0, acc[0]);
    acc[1] = wmma_bf16(a0, b1, acc[1]);
    acc[2] = wmma_bf16(a0, b2, acc[2]);
    acc[3] = wmma_bf16(a0, b3, acc[3]);
    acc[4] = wmma_bf16(a1, b0, acc[4]);
    acc[5] = wmma_bf16(a1, b1, acc[5]);
    acc[6] = wmma_bf16(a1, b2, acc[6]);
    acc[7] = wmma_bf16(a1, b3, acc[7]);
    sched_loads_then_wmma<12, 8>();
  }
  const int ncol = lane & 15, mb = (lane >> 4) << 3;
#pragma unroll
  for (int f = 0; f < 4; ++f) {
    const float bv = bfin[nt * 64 + f * 16 + ncol];
#pragma unroll
    for (int r = 0; r < 8; ++r) {
      OUT[((size_t)b * NSQ + m0 + mb + r) * ND + nt * 64 + f * 16 + ncol] =
          acc[f][r] + bv;
      OUT[((size_t)b * NSQ + m0 + 16 + mb + r) * ND + nt * 64 + f * 16 + ncol] =
          acc[4 + f][r] + bv;
    }
  }
}

// ---------------------------------------------------------------------------
// Host launcher
// ---------------------------------------------------------------------------
extern "C" void kernel_launch(void* const* d_in, const int* in_sizes, int n_in,
                              void* d_out, int out_size, void* d_ws, size_t ws_size,
                              hipStream_t stream) {
  (void)in_sizes; (void)n_in; (void)out_size; (void)ws_size;

  const float* Q  = (const float*)d_in[0];
  const float* K  = (const float*)d_in[1];
  const float* V  = (const float*)d_in[2];
  const unsigned char* pad = (const unsigned char*)d_in[3];
  const float* Wq = (const float*)d_in[4];
  const float* bq = (const float*)d_in[5];
  const float* Wk = (const float*)d_in[6];
  const float* bk = (const float*)d_in[7];
  const float* Wv = (const float*)d_in[8];
  const float* bv = (const float*)d_in[9];
  const float* Wo = (const float*)d_in[10];
  const float* bo = (const float*)d_in[11];
  const float* Wf = (const float*)d_in[12];
  const float* bf = (const float*)d_in[13];
  float* out = (float*)d_out;

  char* base = (char*)d_ws;
  size_t off = 0;
  auto take = [&](size_t bytes) -> char* {
    char* p = base + off;
    off = (off + bytes + 255) & ~(size_t)255;
    return p;
  };

  __bf16* qh  = (__bf16*)take((size_t)NB * NH * NSQ * NDK * 2);  // 16 MB
  __bf16* kh  = (__bf16*)take((size_t)NB * NH * NSK * NDK * 2);  // 16 MB
  __bf16* vh  = (__bf16*)take((size_t)NB * NH * NSK * NDV * 2);  // 16 MB
  __bf16* vhT = (__bf16*)take((size_t)NB * NH * NDV * NSK * 2);  // 16 MB
  __bf16* WqT = (__bf16*)take((size_t)NH * NDK * ND * 2);        //  2 MB
  __bf16* WkT = (__bf16*)take((size_t)NH * NDK * ND * 2);
  __bf16* WvT = (__bf16*)take((size_t)NH * NDV * ND * 2);
  __bf16* WoT = (__bf16*)take((size_t)NH * NDV * NDV * 2);
  __bf16* WfT = (__bf16*)take((size_t)ND * (NH * NDV) * 2);      //  2 MB
  // aliases (producers are dead by the time these are written):
  __bf16* ctx   = vh;   // attention output overlays vh
  __bf16* heads = qh;   // concat overlays qh

  const int thr = 256;
  auto gsz = [](size_t total, int t) {
    size_t g = (total + t - 1) / t;
    return (unsigned)(g > 8192 ? 8192 : g);
  };

  // 1) weight transpose + convert (N-major bf16)
  {
    size_t t = (size_t)NH * ND * NDK;
    k_tcvt_f32<<<gsz(t, thr), thr, 0, stream>>>(Wq, WqT, ND, NDK, t);
    k_tcvt_f32<<<gsz(t, thr), thr, 0, stream>>>(Wk, WkT, ND, NDK, t);
    k_tcvt_f32<<<gsz(t, thr), thr, 0, stream>>>(Wv, WvT, ND, NDV, t);
  }
  {
    size_t t = (size_t)NH * NDV * NDV;
    k_tcvt_f32<<<gsz(t, thr), thr, 0, stream>>>(Wo, WoT, NDV, NDV, t);
  }
  {
    size_t t = (size_t)(NH * NDV) * ND;
    k_tcvt_f32<<<gsz(t, thr), thr, 0, stream>>>(Wf, WfT, NH * NDV, ND, t);
  }

  // 2) QKV projections (WMMA GEMMs, 32x64 tile per wave)
  {
    dim3 grid(NB * NH * (NSQ / 128));
    k_proj<<<grid, 128, 0, stream>>>(Q, WqT, bq, qh, NSQ);
    k_proj<<<grid, 128, 0, stream>>>(K, WkT, bk, kh, NSK);
    k_proj<<<grid, 128, 0, stream>>>(V, WvT, bv, vh, NSK);
  }

  // 3) transpose V heads for contiguous PV B-fragments
  {
    size_t t = (size_t)NB * NH * NSK * NDV;
    k_t_bf16<<<gsz(t, thr), thr, 0, stream>>>(vh, vhT, NSK, NDV, t);
  }

  // 4) flash attention (causal + padding, online softmax, async K staging)
  {
    dim3 grid(NB * NH * (NSQ / 16) / 4);  // 4 waves per block
    k_flash_attn<<<grid, 128, 0, stream>>>(qh, kh, vhT, pad, ctx);
  }

  // 5) per-head Wo into concat layout
  {
    dim3 grid(NB * NH * (NSQ / 64));
    k_wo<<<grid, 128, 0, stream>>>(ctx, WoT, bo, heads);
  }

  // 6) final 1024x1024 projection, fp32 output
  {
    dim3 grid(NB * (NSQ / 128) * (ND / 64));
    k_final<<<grid, 128, 0, stream>>>(heads, WfT, bf, out);
  }
}
